// VSSLayer_up_56538949485142
// MI455X (gfx1250) — matrix operand order repaired and checked by reading, not compiled, not run on hardware
//
#include <hip/hip_runtime.h>
#include <hip/hip_bf16.h>
#include <math.h>

typedef __attribute__((ext_vector_type(16))) _Float16 v16h;
typedef __attribute__((ext_vector_type(8)))  _Float16 v8h;
typedef __attribute__((ext_vector_type(4)))  _Float16 v4h;
typedef __attribute__((ext_vector_type(8)))  float    v8f;
typedef __attribute__((ext_vector_type(4)))  float    v4f;
typedef __attribute__((ext_vector_type(4)))  unsigned int u32x4;
typedef __attribute__((ext_vector_type(8)))  int      i32x8;
typedef __attribute__((ext_vector_type(4)))  int      i32x4;

#define XSTR(x) STR(x)
#define STR(x) #x
#pragma message("CDNA5 build: clang major = " XSTR(__clang_major__))

#if defined(__has_builtin)
#  if __has_builtin(__builtin_amdgcn_tensor_load_to_lds)
#    define HAVE_TDM 1
#    pragma message("CDNA5 build: tensor_load_to_lds builtin AVAILABLE")
#  else
#    warning "CDNA5 build: tensor_load_to_lds builtin NOT available (manual A-tile staging)"
#  endif
#  if __has_builtin(__builtin_amdgcn_s_wait_tensorcnt)
#    define HAVE_WAIT_TENSOR 1
#    pragma message("CDNA5 build: s_wait_tensorcnt builtin AVAILABLE")
#  else
#    warning "CDNA5 build: s_wait_tensorcnt builtin NOT available (inline asm fallback)"
#  endif
#endif

#define NB   4
#define CHN  256
#define HH   64
#define WWD  64
#define HWSZ 4096
#define LLEN 4096
#define DMOD 64
#define DINN 128
#define NSTATE 16
#define DTRANK 4
#define EPSBN 1e-5f

__device__ __forceinline__ float sigmoidf_(float x){ return 1.0f/(1.0f+__expf(-x)); }
__device__ __forceinline__ float siluf_(float x){ return x*sigmoidf_(x); }
__device__ __forceinline__ float actf_(float x,int a){
  if(a==1) return fmaxf(x,0.f);
  if(a==2) return siluf_(x);
  return x;
}

// ---------------------------------------------------------------------------
// WMMA operand loads. LDS row stride 40 halfwords (80B), rows hold contiguous K.
//   A operand lane (hi=lane>>4, m=lane&15): elems 0..7 = K[hi*8..], 8..15 = K[16+hi*8..]
//   B operand lane (hi, n=lane&15):         elems 0..7 = K[hi*8..], 8..15 = K[16+hi*8..]
// Both = two aligned 16B LDS loads.
// ---------------------------------------------------------------------------
__device__ __forceinline__ v16h ld_opA(const _Float16* p, int row, int hi)
{
  const _Float16* pa = p + row*40;
  v8h a0 = *(const v8h*)(pa + hi*8);
  v8h a1 = *(const v8h*)(pa + 16 + hi*8);
  v16h A;
  #pragma unroll
  for (int i=0;i<8;++i){ A[i]=a0[i]; A[i+8]=a1[i]; }
  return A;
}
__device__ __forceinline__ v16h ld_opB(const _Float16* p, int row, int hi)
{
  const _Float16* pb = p + row*40;
  v8h b0 = *(const v8h*)(pb + hi*8);
  v8h b1 = *(const v8h*)(pb + 16 + hi*8);
  v16h B;
  #pragma unroll
  for (int i=0;i<8;++i){ B[i]=b0[i]; B[i+8]=b1[i]; }
  return B;
}
#define WMMA_F16(A,B,C) __builtin_amdgcn_wmma_f32_16x16x32_f16(false,(A),false,(B),(short)0,(C),false,false)

#if defined(HAVE_TDM)
// TDM: load a [nrows x 32] f16 tile (row stride = rowstride elems) from global
// into LDS at lds_off; pad 4 DWORDs after every 16 DWORDs so each 64B row
// lands on an 80B (40-halfword) LDS stride.
__device__ __forceinline__ void tdm_load_tile_f16(const _Float16* gsrc,
                                                  unsigned lds_off,
                                                  int rowlen, int nrows,
                                                  int rowstride)
{
  unsigned long long ga = (unsigned long long)(size_t)gsrc;
  u32x4 g0; i32x8 g1; i32x4 g2 = {0,0,0,0}; i32x4 g3 = {0,0,0,0};
  g0[0] = 1u;                                       // count=1 user descriptor
  g0[1] = lds_off;                                  // lds_addr
  g0[2] = (unsigned)(ga & 0xffffffffu);             // global_addr[31:0]
  g0[3] = (unsigned)((ga >> 32) & 0x01ffffffu)      // global_addr[56:32]
        | (2u << 30);                               // type = 2 (image)
  unsigned dim0 = (unsigned)rowstride;              // tensor_dim0
  unsigned dim1 = (unsigned)nrows;                  // tensor_dim1
  g1[0] = (int)((1u << 16)                          // data_size = 2 bytes
        |       (1u << 20)                          // pad_enable
        |       (3u << 22)                          // pad_interval = 16 dwords
        |       (3u << 25));                        // pad_amount   = 4 dwords
  g1[1] = (int)((dim0 & 0xffffu) << 16);            // tensor_dim0[15:0]
  g1[2] = (int)((dim0 >> 16) | ((dim1 & 0xffffu) << 16));
  g1[3] = (int)((dim1 >> 16) | ((unsigned)rowlen << 16));  // tile_dim0
  g1[4] = (int)(unsigned)nrows;                     // tile_dim1
  g1[5] = (int)(unsigned)rowstride;                 // tensor_dim0_stride lo
  g1[6] = 0;
  g1[7] = 0;
#if __clang_major__ >= 23
  i32x8 g4 = {0,0,0,0,0,0,0,0};
  __builtin_amdgcn_tensor_load_to_lds(g0, g1, g2, g3, g4, 0);
#else
  __builtin_amdgcn_tensor_load_to_lds(g0, g1, g2, g3, 0);
#endif
}
#endif

__device__ __forceinline__ void wait_tensor0()
{
#if defined(HAVE_WAIT_TENSOR)
  __builtin_amdgcn_s_wait_tensorcnt(0);
#else
  asm volatile("s_wait_tensorcnt 0x0" ::: "memory");
#endif
}

// ---------------------------------------------------------------------------
// Pointwise 1x1 conv as implicit GEMM: Y[n,co,t] = act(BN(W.X + bias) + res)
// Block tile 64co x 64t, 8 waves = 2(co-pair) x 4(t); each wave computes two
// 16x16 tiles sharing one B operand (2 back-to-back WMMAs). LDS double-
// buffered, one barrier per 32-wide K step. Ci, Co multiples of 32.
// ---------------------------------------------------------------------------
__global__ __launch_bounds__(256)
void k_conv1x1(const float* __restrict__ X, int XCn, int ci_off,
               const float* __restrict__ Wt, int Ci, int Co,
               const float* __restrict__ bias,
               const float* __restrict__ bng, const float* __restrict__ bnb,
               const float* __restrict__ bnm, const float* __restrict__ bnv,
               const float* __restrict__ res, int resCn, int res_off,
               float* __restrict__ Y, int YCn, int co_off, int act)
{
  __shared__ __align__(16) _Float16 lA[2][64*40];
  __shared__ __align__(16) _Float16 lB[2][64*40];
  const int tid = threadIdx.x;
  const int t0  = blockIdx.x*64;
  const int cog = blockIdx.y*64;
  const int n   = blockIdx.z;
  const int lane = tid & 31, wv = tid >> 5;
  const int ca = wv & 1, bt = wv >> 1;       // co pair / t tile
  const int hi = lane >> 4, ln = lane & 15;

  const int arow = tid >> 2, aq = (tid & 3)*8;        // A staging: 64 x 32
  const int ci2 = (tid & 15)*2, tq = (tid >> 4)*4;    // B staging (ci pairs)

  auto stage = [&](int p, int ci0){
    { // A: 64co x 32ci, 8 f32 -> 8 f16 (16B ds store)
      const float* sw = Wt + (size_t)(cog+arow)*Ci + ci0 + aq;
      v4f w0 = *(const v4f*)sw;
      v4f w1 = *(const v4f*)(sw+4);
      _Float16* dst = &lA[p][arow*40 + aq];
      #pragma unroll
      for (int i=0;i<4;++i){ dst[i]=(_Float16)w0[i]; dst[4+i]=(_Float16)w1[i]; }
    }
    { // B transposed: ci-pair x t-quad -> packed b32 ds stores
      const float* s0 = X + ((size_t)n*XCn + ci_off + ci0 + ci2)*HWSZ + t0 + tq;
      const float* s1 = s0 + HWSZ;
      v4f a = *(const v4f*)s0;
      v4f b = *(const v4f*)s1;
      #pragma unroll
      for (int i=0;i<4;++i){
        _Float16* d = &lB[p][(tq+i)*40 + ci2];
        d[0] = (_Float16)a[i];
        d[1] = (_Float16)b[i];
      }
      if (ci0 + 64 < Ci)
        __builtin_prefetch(s0 + (size_t)64*HWSZ, 0, 0);
    }
  };

  v8f acc0 = {}, acc1 = {};
  const int ksteps = Ci >> 5;
  stage(0, 0);
  __syncthreads();
  for (int kb = 0; kb < ksteps; ++kb) {
    const int p = kb & 1;
    if (kb+1 < ksteps) stage(p^1, (kb+1)*32);
    v16h Bop = ld_opB(lB[p], bt*16 + ln, hi);
    v16h A0  = ld_opA(lA[p], ca*32 + ln, hi);
    v16h A1  = ld_opA(lA[p], ca*32 + 16 + ln, hi);
    acc0 = WMMA_F16(A0, Bop, acc0);
    acc1 = WMMA_F16(A1, Bop, acc1);
    __syncthreads();
  }

  const int t = t0 + bt*16 + ln;
  #pragma unroll
  for (int s=0;s<2;++s){
    v8f a = s ? acc1 : acc0;
    #pragma unroll
    for (int r=0;r<8;++r){
      int m  = r + 8*hi;            // C/D layout: VGPR r -> M = r + 8*(lane>>4)
      int co = cog + ca*32 + s*16 + m;
      float v = a[r];
      if (bias) v += bias[co];
      if (bng){
        float sc = bng[co]*__frsqrt_rn(bnv[co]+EPSBN);
        v = (v - bnm[co])*sc + bnb[co];
      }
      if (res) v += res[((size_t)n*resCn + res_off + co)*HWSZ + t];
      Y[((size_t)n*YCn + co_off + co)*HWSZ + t] = actf_(v, act);
    }
  }
}

// ---------------------------------------------------------------------------
// Row-major token GEMM: Y[M,Nld] = A[M,Kld](f16) x B[K,N](f32->f16)
// Block tile 64M x 64N; each wave = two 16x16 tiles sharing one B operand.
// A tile (64x32 f16) fetched by the Tensor Data Mover. LDS double-buffered.
// K, M multiples of 32 / 64.
// ---------------------------------------------------------------------------
__global__ __launch_bounds__(256)
void k_gemm_rm(const _Float16* __restrict__ A, int Kld,
               const float* __restrict__ Bw, int K, int N,
               float* __restrict__ Y, int Nld)
{
  __shared__ __align__(16) _Float16 lA[2][64*40];
  __shared__ __align__(16) _Float16 lB[2][64*40];
  const int tid = threadIdx.x;
  const int m0 = blockIdx.x*64;
  const int n0 = blockIdx.y*64;
  const int lane = tid & 31, wv = tid >> 5;
  const int mr = wv & 1, nt = wv >> 1;       // M pair / N tile
  const int hi = lane >> 4, ln = lane & 15;
  const bool fullN = (n0 + 64) <= N;
  const int k2 = (tid & 15)*2, n4 = (tid >> 4)*4;

  auto stage = [&](int p, int k0){
#if defined(HAVE_TDM)
    if (wv == 0) {   // one wave issues the DMA; TENSORcnt tracks completion
      tdm_load_tile_f16(A + (size_t)m0*Kld + k0, (unsigned)(size_t)&lA[p][0],
                        32, 64, Kld);
      wait_tensor0();
    }
#else
    { // fallback: manual A tile copy (16B per thread)
      int row = tid >> 2, kq = (tid & 3)*8;
      *(v8h*)&lA[p][row*40 + kq] =
          *(const v8h*)(A + (size_t)(m0+row)*Kld + k0 + kq);
    }
#endif
    if (fullN) { // clean vector path (N=256, N=64): k-pair x n-quad packing
      const float* s0 = Bw + (size_t)(k0+k2)*N + n0 + n4;
      const float* s1 = s0 + N;
      v4f a = *(const v4f*)s0;
      v4f b = *(const v4f*)s1;
      #pragma unroll
      for (int i=0;i<4;++i){
        _Float16* d = &lB[p][(n4+i)*40 + k2];
        d[0] = (_Float16)a[i];
        d[1] = (_Float16)b[i];
      }
    } else {     // guarded path (only the N=36 dbc GEMM)
      int k = tid & 31, n8 = (tid >> 5)*8;
      #pragma unroll
      for (int i=0;i<8;++i){
        int nn = n0 + n8 + i;
        float v = (nn < N) ? Bw[(size_t)(k0+k)*N + nn] : 0.f;
        lB[p][(n8+i)*40 + k] = (_Float16)v;
      }
    }
  };

  v8f acc0 = {}, acc1 = {};
  stage(0, 0);
  __syncthreads();
  const int ksteps = K >> 5;
  for (int kb = 0; kb < ksteps; ++kb) {
    const int p = kb & 1;
    if (kb+1 < ksteps) stage(p^1, (kb+1)*32);
    v16h Bop = ld_opB(lB[p], nt*16 + ln, hi);
    v16h A0  = ld_opA(lA[p], mr*32 + ln, hi);
    v16h A1  = ld_opA(lA[p], mr*32 + 16 + ln, hi);
    acc0 = WMMA_F16(A0, Bop, acc0);
    acc1 = WMMA_F16(A1, Bop, acc1);
    __syncthreads();
  }

  const int nn = n0 + nt*16 + ln;
  if (nn < N){
    #pragma unroll
    for (int s=0;s<2;++s){
      v8f a = s ? acc1 : acc0;
      #pragma unroll
      for (int r=0;r<8;++r){
        int m = m0 + mr*32 + s*16 + r + 8*hi;
        Y[(size_t)m*Nld + nn] = a[r];
      }
    }
  }
}

// ----------------------- memory-bound fp32 kernels -------------------------

__global__ __launch_bounds__(256)
void k_dw3x3_bn(const float* __restrict__ X, const float* __restrict__ dw,
                const float* __restrict__ bng,const float* __restrict__ bnb,
                const float* __restrict__ bnm,const float* __restrict__ bnv,
                float* __restrict__ Y, int dil)
{
  size_t id = (size_t)blockIdx.x*blockDim.x + threadIdx.x;
  if (id >= (size_t)NB*CHN*HWSZ) return;
  int x = id % WWD; int y = (id / WWD) % HH;
  int c = (id / HWSZ) % CHN;
  int n = id / ((size_t)CHN*HWSZ);
  const float* Xc = X + ((size_t)n*CHN + c)*HWSZ;
  const float* w  = dw + c*9;
  float s = 0.f;
  #pragma unroll
  for (int ky=0;ky<3;++ky){
    int yy = y + (ky-1)*dil;
    if (yy<0||yy>=HH) continue;
    #pragma unroll
    for (int kx=0;kx<3;++kx){
      int xx = x + (kx-1)*dil;
      if (xx<0||xx>=WWD) continue;
      s += w[ky*3+kx]*Xc[yy*WWD+xx];
    }
  }
  float sc = bng[c]*__frsqrt_rn(bnv[c]+EPSBN);
  Y[id] = (s - bnm[c])*sc + bnb[c];
}

__global__ __launch_bounds__(256)
void k_add(const float* __restrict__ A, const float* __restrict__ B,
           float* __restrict__ Y, size_t n)
{
  size_t id = (size_t)blockIdx.x*blockDim.x + threadIdx.x;
  if (id < n) Y[id] = A[id] + B[id];
}

__global__ __launch_bounds__(256)
void k_ca_reduce(const float* __restrict__ X, float* __restrict__ avg,
                 float* __restrict__ mx)
{
  int w = (blockIdx.x*blockDim.x + threadIdx.x) >> 5;
  int lane = threadIdx.x & 31;
  if (w >= NB*CHN) return;
  const float* p = X + (size_t)w*HWSZ;
  float s = 0.f, m = -1e30f;
  for (int i=lane;i<HWSZ;i+=32){ float v=p[i]; s+=v; m=fmaxf(m,v); }
  for (int off=16;off>=1;off>>=1){
    s += __shfl_xor(s,off,32);
    m  = fmaxf(m,__shfl_xor(m,off,32));
  }
  if (lane==0){ avg[w]=s*(1.f/HWSZ); mx[w]=m; }
}

__global__ __launch_bounds__(256)
void k_ca_mlp(const float* __restrict__ avg, const float* __restrict__ mx,
              const float* __restrict__ w1, const float* __restrict__ b1,
              const float* __restrict__ w2, const float* __restrict__ b2,
              float* __restrict__ att)
{
  __shared__ float sa[CHN], sm[CHN], ha[16], hm[16];
  int n = blockIdx.x, tid = threadIdx.x;
  sa[tid] = avg[n*CHN + tid]; sm[tid] = mx[n*CHN + tid];
  __syncthreads();
  if (tid < 16){
    float xa = b1[tid], xm = b1[tid];
    for (int c=0;c<CHN;++c){ float w=w1[tid*CHN+c]; xa+=w*sa[c]; xm+=w*sm[c]; }
    ha[tid]=fmaxf(xa,0.f); hm[tid]=fmaxf(xm,0.f);
  }
  __syncthreads();
  float oa = b2[tid], om = b2[tid];
  for (int j=0;j<16;++j){ float w=w2[tid*16+j]; oa+=w*ha[j]; om+=w*hm[j]; }
  att[n*CHN+tid] = sigmoidf_(oa+om);
}

__global__ __launch_bounds__(256)
void k_ca_apply(const float* __restrict__ X, const float* __restrict__ att,
                const float* __restrict__ B1, float* __restrict__ Y)
{
  size_t id = (size_t)blockIdx.x*blockDim.x + threadIdx.x;
  if (id >= (size_t)NB*CHN*HWSZ) return;
  size_t nc = id / HWSZ;
  Y[id] = X[id]*att[nc] + B1[id];
}

__global__ __launch_bounds__(256)
void k_sim(const float* __restrict__ G, const float* __restrict__ Lc,
           float* __restrict__ simo)
{
  int w = (blockIdx.x*blockDim.x + threadIdx.x) >> 5;
  int lane = threadIdx.x & 31;
  if (w >= NB*HWSZ) return;
  int n = w / HWSZ, p = w % HWSZ;
  float dot=0.f, ng=0.f, nl=0.f;
  for (int c=lane;c<CHN;c+=32){
    float g = G[((size_t)n*CHN + c)*HWSZ + p];
    float l = Lc[((size_t)n*CHN + c)*HWSZ + p];
    dot += g*l; ng += g*g; nl += l*l;
  }
  for (int off=16;off>=1;off>>=1){
    dot += __shfl_xor(dot,off,32);
    ng  += __shfl_xor(ng,off,32);
    nl  += __shfl_xor(nl,off,32);
  }
  if (lane==0)
    simo[w] = sigmoidf_(dot/(fmaxf(sqrtf(ng),1e-8f)*fmaxf(sqrtf(nl),1e-8f)));
}

__global__ __launch_bounds__(256)
void k_scale_sim(const float* __restrict__ X, const float* __restrict__ simo,
                 float* __restrict__ Y)
{
  size_t id = (size_t)blockIdx.x*blockDim.x + threadIdx.x;
  if (id >= (size_t)NB*CHN*HWSZ) return;
  size_t p = id % HWSZ;
  size_t n = id / ((size_t)CHN*HWSZ);
  Y[id] = X[id]*simo[n*HWSZ + p];
}

__global__ __launch_bounds__(256)
void k_up2x(const float* __restrict__ X, float* __restrict__ Y)
{
  size_t id = (size_t)blockIdx.x*blockDim.x + threadIdx.x;
  if (id >= (size_t)NB*CHN*128*128) return;
  int x = id % 128; int y = (id/128) % 128;
  size_t nc = id / (128*128);
  const float sc = 63.f/127.f;          // align_corners=True
  float fy = y*sc, fx = x*sc;
  int y0 = (int)fy, x0 = (int)fx;
  int y1 = min(y0+1, HH-1), x1 = min(x0+1, WWD-1);
  float wy = fy - y0, wx = fx - x0;
  const float* p = X + nc*HWSZ;
  float v = (p[y0*WWD+x0]*(1.f-wx)+p[y0*WWD+x1]*wx)*(1.f-wy)
          + (p[y1*WWD+x0]*(1.f-wx)+p[y1*WWD+x1]*wx)*wy;
  Y[id] = v;
}

// ------------------------------ Mamba kernels ------------------------------

__global__ __launch_bounds__(256)
void k_mamba_ln(const float* __restrict__ G, int c0,
                const float* __restrict__ g, const float* __restrict__ b,
                _Float16* __restrict__ Hh)
{
  int w = (blockIdx.x*blockDim.x + threadIdx.x) >> 5;   // one wave per token
  int lane = threadIdx.x & 31;
  if (w >= NB*LLEN) return;
  int bn = w / LLEN, l = w % LLEN;
  float v0 = G[((size_t)bn*CHN + c0 + lane)*HWSZ + l];
  float v1 = G[((size_t)bn*CHN + c0 + lane + 32)*HWSZ + l];
  float s = v0+v1, sq = v0*v0+v1*v1;
  for (int off=16;off>=1;off>>=1){
    s += __shfl_xor(s,off,32); sq += __shfl_xor(sq,off,32);
  }
  float mu  = s*(1.f/DMOD);
  float var = sq*(1.f/DMOD) - mu*mu;
  float inv = __frsqrt_rn(var + 1e-6f);
  Hh[(size_t)w*DMOD + lane]      = (_Float16)((v0-mu)*inv*g[lane]    + b[lane]);
  Hh[(size_t)w*DMOD + lane + 32] = (_Float16)((v1-mu)*inv*g[lane+32] + b[lane+32]);
}

__global__ __launch_bounds__(256)
void k_mamba_conv(const float* __restrict__ XZ, const float* __restrict__ cw,
                  const float* __restrict__ cb,
                  float* __restrict__ xc, _Float16* __restrict__ xch)
{
  size_t id = (size_t)blockIdx.x*blockDim.x + threadIdx.x;
  if (id >= (size_t)NB*LLEN*DINN) return;
  int d = id % DINN; size_t t = id / DINN;
  int l = t % LLEN; size_t tb = t - l;
  float s = cb[d];
  #pragma unroll
  for (int k=0;k<4;++k){
    int ls = l - 3 + k;
    if (ls >= 0) s += cw[d*4+k]*XZ[(tb+ls)*(2*DINN) + d];
  }
  float v = siluf_(s);
  xc[id] = v; xch[id] = (_Float16)v;
}

__global__ __launch_bounds__(256)
void k_mamba_dt(const float* __restrict__ dbc, const float* __restrict__ wdt,
                const float* __restrict__ bdt, float* __restrict__ dt)
{
  size_t id = (size_t)blockIdx.x*blockDim.x + threadIdx.x;
  if (id >= (size_t)NB*LLEN*DINN) return;
  int d = id % DINN; size_t t = id / DINN;
  float s = bdt[d];
  #pragma unroll
  for (int r=0;r<DTRANK;++r) s += dbc[t*48 + r]*wdt[r*DINN + d];
  dt[id] = (s > 20.f) ? s : log1pf(__expf(s));
}

// selective scan: 16 lanes = 16 states per (batch, channel); 2 channels/wave
__global__ __launch_bounds__(256)
void k_mamba_scan(const float* __restrict__ dt, const float* __restrict__ dbc,
                  const float* __restrict__ xc, const float* __restrict__ Alog,
                  const float* __restrict__ Dp, float* __restrict__ yb)
{
  int w = (blockIdx.x*blockDim.x + threadIdx.x) >> 5;
  int lane = threadIdx.x & 31;
  int hi = lane >> 4, n = lane & 15;
  if (w >= NB*(DINN/2)) return;
  int b = w / (DINN/2);
  int d = (w % (DINN/2))*2 + hi;
  float An = -__expf(Alog[d*NSTATE + n]);
  float Dd = Dp[d];
  float h = 0.f;
  size_t tb = (size_t)b*LLEN;
  for (int l=0;l<LLEN;++l){
    size_t t = tb + l;
    float dtv = dt[t*DINN + d];
    float xv  = xc[t*DINN + d];
    float Bn  = dbc[t*48 + DTRANK + n];
    float Cn  = dbc[t*48 + DTRANK + NSTATE + n];
    h = __expf(dtv*An)*h + dtv*Bn*xv;
    float p = h*Cn;
    for (int off=8;off>=1;off>>=1) p += __shfl_xor(p,off,16);
    if (n==0) yb[t*DINN + d] = p + Dd*xv;
  }
}

__global__ __launch_bounds__(256)
void k_mamba_gate(const float* __restrict__ yb, const float* __restrict__ XZ,
                  _Float16* __restrict__ ygh)
{
  size_t id = (size_t)blockIdx.x*blockDim.x + threadIdx.x;
  if (id >= (size_t)NB*LLEN*DINN) return;
  int d = id % DINN; size_t t = id / DINN;
  float z = XZ[t*(2*DINN) + DINN + d];
  ygh[id] = (_Float16)(yb[id]*siluf_(z));
}

__global__ __launch_bounds__(256)
void k_tok_add(const float* __restrict__ outg, float* __restrict__ G, int c0)
{
  size_t id = (size_t)blockIdx.x*blockDim.x + threadIdx.x;
  if (id >= (size_t)NB*LLEN*DMOD) return;
  int ch = id % DMOD; size_t t = id / DMOD;
  int b = t / LLEN, l = t % LLEN;
  size_t gi = ((size_t)b*CHN + c0 + ch)*HWSZ + l;
  G[gi] += outg[id];
}

// ------------------------------- host side ---------------------------------

#define INF(i) ((const float*)d_in[(i)])
static const size_t TSZ = (size_t)NB*CHN*HWSZ;          // elems per 256ch tensor
static const int    EB  = (int)((TSZ + 255)/256);       // elemwise grid

static void run_conv(const float* X,int XCn,int ci_off,const float* Wt,int Ci,
                     const float* bias,const float* g,const float* b,
                     const float* m,const float* v,
                     const float* res,int resCn,int res_off,
                     float* Y,int YCn,int co_off,int act,hipStream_t s)
{
  dim3 grid(HWSZ/64, CHN/64, NB);
  k_conv1x1<<<grid,256,0,s>>>(X,XCn,ci_off,Wt,Ci,CHN,bias,g,b,m,v,
                              res,resCn,res_off,Y,YCn,co_off,act);
}

static void run_ib(const float* Xin,int XCn,int ci_off,
                   const float* resp,int resCn,int res_off,
                   float* Yout,int YCn,int co_off,int rate,
                   void* const* d_in,int base,
                   float* t1,float* t2,hipStream_t s)
{
  run_conv(Xin,XCn,ci_off, INF(base),CHN, nullptr,
           INF(base+1),INF(base+2),INF(base+3),INF(base+4),
           nullptr,0,0, t1,CHN,0, 1, s);
  k_dw3x3_bn<<<EB,256,0,s>>>(t1, INF(base+5),
           INF(base+6),INF(base+7),INF(base+8),INF(base+9), t2, rate);
  run_conv(t2,CHN,0, INF(base+10),CHN, nullptr,
           nullptr,nullptr,nullptr,nullptr, nullptr,0,0, t1,CHN,0, 2, s);
  run_conv(t1,CHN,0, INF(base+11),CHN, nullptr,
           INF(base+12),INF(base+13),INF(base+14),INF(base+15),
           resp,resCn,res_off, Yout,YCn,co_off, 1, s);
}

static void run_mamba(void* const* d_in,int base,int c0,float* G,
                      _Float16* Hh,float* XZ,float* XC,_Float16* XCh,
                      float* DBC,float* DT,float* YB,_Float16* YGh,float* OUT,
                      hipStream_t s)
{
  const int TOK = NB*LLEN;
  k_mamba_ln<<<TOK/8,256,0,s>>>(G, c0, INF(base), INF(base+1), Hh);
  k_gemm_rm<<<dim3(TOK/64, (2*DINN)/64),256,0,s>>>(Hh, DMOD, INF(base+2),
                                                   DMOD, 2*DINN, XZ, 2*DINN);
  int EM = (TOK*DINN + 255)/256;
  k_mamba_conv<<<EM,256,0,s>>>(XZ, INF(base+3), INF(base+4), XC, XCh);
  k_gemm_rm<<<dim3(TOK/64, 1),256,0,s>>>(XCh, DINN, INF(base+5),
                                         DINN, DTRANK+2*NSTATE, DBC, 48);
  k_mamba_dt<<<EM,256,0,s>>>(DBC, INF(base+6), INF(base+7), DT);
  k_mamba_scan<<<(NB*(DINN/2))/8,256,0,s>>>(DT, DBC, XC, INF(base+8),
                                            INF(base+9), YB);
  k_mamba_gate<<<EM,256,0,s>>>(YB, XZ, YGh);
  k_gemm_rm<<<dim3(TOK/64, 1),256,0,s>>>(YGh, DINN, INF(base+10),
                                         DINN, DMOD, OUT, DMOD);
  k_tok_add<<<(TOK*DMOD+255)/256,256,0,s>>>(OUT, G, c0);
}

extern "C" void kernel_launch(void* const* d_in, const int* in_sizes, int n_in,
                              void* d_out, int out_size, void* d_ws, size_t ws_size,
                              hipStream_t stream)
{
  char* ws = (char*)d_ws;
  const size_t MB = 1ull<<20;
  float*     bufA  = (float*)(ws +   0*MB);
  float*     bufB  = (float*)(ws +  16*MB);
  float*     bufC  = (float*)(ws +  32*MB);
  float*     bufD  = (float*)(ws +  48*MB);
  float*     bufE  = (float*)(ws +  64*MB);
  float*     bufG  = (float*)(ws +  80*MB);
  float*     bufCC = (float*)(ws +  96*MB);   // 48 MB concat [b2|b3|b4]
  _Float16*  mHh   = (_Float16*)(ws + 144*MB);
  float*     mXZ   = (float*)(ws + 146*MB);
  float*     mXC   = (float*)(ws + 162*MB);
  _Float16*  mXCh  = (_Float16*)(ws + 170*MB);
  float*     mDBC  = (float*)(ws + 174*MB);
  float*     mDT   = (float*)(ws + 177*MB);
  float*     mYB   = (float*)(ws + 185*MB);
  _Float16*  mYGh  = (_Float16*)(ws + 193*MB);
  float*     mOut  = (float*)(ws + 197*MB);
  float*     cAvg  = (float*)(ws + 201*MB);
  float*     cMax  = cAvg + NB*CHN;
  float*     cAtt  = cMax + NB*CHN;
  float*     simb  = cAtt + NB*CHN;

  const float* x = INF(0);

  // ---------------- local feature branch ----------------
  run_conv(x,CHN,0, INF(1),CHN, INF(2), INF(3),INF(4),INF(5),INF(6),
           nullptr,0,0, bufA,CHN,0, 2, stream);
  k_dw3x3_bn<<<EB,256,0,stream>>>(x, INF(7), INF(8),INF(9),INF(10),INF(11),
                                  bufB, 1);
  run_conv(bufB,CHN,0, INF(12),CHN, nullptr, nullptr,nullptr,nullptr,nullptr,
           nullptr,0,0, bufC,CHN,0, 2, stream);
  k_add<<<EB,256,0,stream>>>(bufA, bufC, bufD, TSZ);
  run_ib(bufD,CHN,0,   bufD,CHN,0,     bufCC,3*CHN,0,     1, d_in,13, bufA,bufB, stream);
  run_ib(bufCC,3*CHN,0,   bufCC,3*CHN,0,   bufCC,3*CHN,CHN,   2, d_in,29, bufA,bufB, stream);
  run_ib(bufCC,3*CHN,CHN, bufCC,3*CHN,CHN, bufCC,3*CHN,2*CHN, 3, d_in,45, bufA,bufB, stream);
  run_conv(bufCC,3*CHN,0, INF(61),3*CHN, INF(62),
           nullptr,nullptr,nullptr,nullptr, nullptr,0,0, bufB,CHN,0, 0, stream);
  k_ca_reduce<<<(NB*CHN)/8,256,0,stream>>>(bufB, cAvg, cMax);
  k_ca_mlp<<<NB,256,0,stream>>>(cAvg, cMax, INF(63),INF(64),INF(65),INF(66), cAtt);
  k_ca_apply<<<EB,256,0,stream>>>(bufB, cAtt, bufD, bufC);   // local -> bufC

  // ---------------- global branch (VSS blocks) ----------------
  hipMemcpyAsync(bufG, x, TSZ*sizeof(float), hipMemcpyDeviceToDevice, stream);
  for (int d = 0; d < 2; ++d){
    int base = 67 + 11*d;
    for (int c = 0; c < 4; ++c)
      run_mamba(d_in, base, c*DMOD, bufG,
                mHh, mXZ, mXC, mXCh, mDBC, mDT, mYB, mYGh, mOut, stream);
  }

  // ---------------- feature embed + residual + upsample ----------------
  k_add<<<EB,256,0,stream>>>(bufG, bufC, bufA, TSZ);          // g + l
  k_dw3x3_bn<<<EB,256,0,stream>>>(bufA, INF(89),
                                  INF(90),INF(91),INF(92),INF(93), bufB, 1);
  run_conv(bufB,CHN,0, INF(94),CHN, nullptr,
           nullptr,nullptr,nullptr,nullptr, nullptr,0,0, bufE,CHN,0, 2, stream);
  k_sim<<<(NB*HWSZ)/8,256,0,stream>>>(bufG, bufC, simb);
  k_scale_sim<<<EB,256,0,stream>>>(bufE, simb, bufA);
  k_dw3x3_bn<<<EB,256,0,stream>>>(bufA, INF(95),
                                  INF(96),INF(97),INF(98),INF(99), bufB, 1);
  run_conv(bufB,CHN,0, INF(100),CHN, nullptr,
           nullptr,nullptr,nullptr,nullptr, nullptr,0,0, bufD,CHN,0, 2, stream);
  k_add<<<EB,256,0,stream>>>(bufD, bufE, bufA, TSZ);
  k_add<<<EB,256,0,stream>>>(bufA, bufG, bufB, TSZ);
  size_t OUTN = (size_t)NB*CHN*128*128;
  k_up2x<<<(int)((OUTN+255)/256),256,0,stream>>>(bufB, (float*)d_out);
}
